// PointNetDenseMap_87668872446782
// MI455X (gfx1250) — matrix operand order repaired and checked by reading, not compiled, not run on hardware
//
#include <hip/hip_runtime.h>
#include <hip/hip_bf16.h>
#include <math.h>

// ============================================================================
// PointNet dense-seg forward for MI455X (gfx1250, wave32, WMMA).
// All 1x1 convs run through a tiled v_wmma_f32_16x16x32_f16 GEMM with a fused
// epilogue that accumulates BN batch statistics (sum/sumsq per channel) and,
// for the max-pooled 1024-ch layers, per-(b,c) max/min so the (16,1024,4096)
// tensors are never materialized (saves ~540MB of HBM traffic @ 23.3 TB/s).
// GEMM is template-specialized: the 6 heavy interior layers run an unguarded
// straight-line loader (no exec-mask divergence in the K-loop); only the
// Cin=3 layers and the 1088-ch concat layer take the guarded path.
// LDS is double-buffered: one barrier per K-step, loads of tile t+1 overlap
// the WMMAs of tile t.
// ============================================================================

typedef __attribute__((ext_vector_type(16))) _Float16 v16h;
typedef __attribute__((ext_vector_type(8)))  float    v8f;

constexpr int BATCH = 16;
constexpr int NPTS  = 4096;
constexpr int NFACE = 8192;

// ---------------- d_out layout (float offsets), tuple order:
// (area, cos, xo, logp, xo, area, cos)
constexpr size_t OUT_AREA1 = 0;
constexpr size_t OUT_COS1  = OUT_AREA1 + (size_t)BATCH * NFACE;
constexpr size_t OUT_XO1   = OUT_COS1  + (size_t)BATCH * NFACE * 3;
constexpr size_t OUT_LOGP  = OUT_XO1   + (size_t)BATCH * NPTS * 2;
constexpr size_t OUT_XO2   = OUT_LOGP  + (size_t)BATCH * NPTS * 2;
constexpr size_t OUT_AREA2 = OUT_XO2   + (size_t)BATCH * NPTS * 2;
constexpr size_t OUT_COS2  = OUT_AREA2 + (size_t)BATCH * NFACE;

// ---------------- workspace layout (float offsets) ----------------
constexpr size_t OFF_SUM   = 0;            // [12 slots][1024] BN sum accum
constexpr size_t OFF_SQ    = 12 * 1024;    // [12][1024] BN sumsq accum
constexpr size_t OFF_SCALE = 24 * 1024;    // [12][1024] BN scale
constexpr size_t OFF_SHIFT = 36 * 1024;    // [12][1024] BN shift
constexpr size_t OFF_MAX   = 48 * 1024;    // [2 red slots][16][1024]
constexpr size_t OFF_MIN   = 80 * 1024;    // [2][16][1024]
constexpr size_t OFF_HMAX  = 112 * 1024;   // stn pooled (16,1024)
constexpr size_t OFF_FC1   = 128 * 1024;   // (16,512)
constexpr size_t OFF_FC2   = 136 * 1024;   // (16,256)
constexpr size_t OFF_T9    = 140 * 1024;   // (16,9) transform
constexpr size_t OFF_G     = 141 * 1024;   // feat global feature (16,1024)
constexpr size_t OFF_XOWS  = 157 * 1024;   // xo (16,4096,2)
constexpr size_t OFF_WF16  = 294912;       // f16 weight pool (as float offset)
constexpr size_t OFF_XT    = 819200;       // transformed pts (16,3,4096)
constexpr size_t OFF_B64   = 1048576;      // (16,64,4096)   stn a1 / pointfeat
constexpr size_t OFF_B128  = 5242880;      // (16,128,4096)  stn a2 / feat a2
constexpr size_t OFF_BE    = 13631488;     // (16,512,4096)  head b1; reused b3
constexpr size_t OFF_BF    = 47185920;     // (16,256,4096)  head b2
// total ~63.96M floats = ~256 MB

// f16 weight pool sub-offsets (in f16 elements), Cin padded to mult of 32
constexpr size_t WO_STN_C1  = 0;
constexpr size_t WO_STN_C2  = WO_STN_C1  + 64 * 32;
constexpr size_t WO_STN_C3  = WO_STN_C2  + 128 * 64;
constexpr size_t WO_FEAT_C1 = WO_STN_C3  + 1024 * 128;
constexpr size_t WO_FEAT_C2 = WO_FEAT_C1 + 64 * 32;
constexpr size_t WO_FEAT_C3 = WO_FEAT_C2 + 128 * 64;
constexpr size_t WO_HEAD_C1 = WO_FEAT_C3 + 1024 * 128;
constexpr size_t WO_HEAD_C2 = WO_HEAD_C1 + 512 * 1088;
constexpr size_t WO_HEAD_C3 = WO_HEAD_C2 + 256 * 512;

// ---------------- d_in leaf order ----------------
// setup_inputs dict order: x, faces, edges, then params flattened as jax tree
// leaves (dicts sorted by key at every level: feat < head < stn; bn* < c* <
// fc*; within {'w','b'}: 'b' < 'w'; within {'g','b'}: 'b' < 'g').
enum {
  IN_X = 0, IN_FACES = 1, IN_EDGES = 2,
  P_FEAT_BN1_B, P_FEAT_BN1_G, P_FEAT_BN2_B, P_FEAT_BN2_G,
  P_FEAT_BN3_B, P_FEAT_BN3_G,
  P_FEAT_C1_B, P_FEAT_C1_W, P_FEAT_C2_B, P_FEAT_C2_W, P_FEAT_C3_B, P_FEAT_C3_W,
  P_HEAD_BN1_B, P_HEAD_BN1_G, P_HEAD_BN2_B, P_HEAD_BN2_G,
  P_HEAD_BN3_B, P_HEAD_BN3_G,
  P_HEAD_C1_B, P_HEAD_C1_W, P_HEAD_C2_B, P_HEAD_C2_W, P_HEAD_C3_B, P_HEAD_C3_W,
  P_HEAD_C4_B, P_HEAD_C4_W, P_HEAD_C5_B, P_HEAD_C5_W,
  P_STN_BN1_B, P_STN_BN1_G, P_STN_BN2_B, P_STN_BN2_G,
  P_STN_BN3_B, P_STN_BN3_G, P_STN_BN4_B, P_STN_BN4_G,
  P_STN_BN5_B, P_STN_BN5_G,
  P_STN_C1_B, P_STN_C1_W, P_STN_C2_B, P_STN_C2_W, P_STN_C3_B, P_STN_C3_W,
  P_STN_FC1_B, P_STN_FC1_W, P_STN_FC2_B, P_STN_FC2_W, P_STN_FC3_B, P_STN_FC3_W
};

// ============================================================================
// WMMA GEMM:   out[b, o, n] = bias[o] + sum_i W[o,i] * in[b, i, n]
// Block: 256 threads (8 waves). Block tile 64(M=Cout) x 128(N=points).
// Wave (2x4 grid): wave tile 16 x 64 -> 4 f32 16x16 accumulators.
// K staged in steps of 32 through double-buffered LDS (f32 -> f16 convert).
// GUARD=false: Cin==CinPad, no concat -> straight-line loader.
// GUARD=true : zero-pad rows >= Cin; rows < splitK broadcast from bcast[b][k].
// ============================================================================
union FragAB { v16h v; uint4 u[2]; };
union H4     { _Float16 h[4]; uint2 u; };

template <bool GUARD>
__global__ __launch_bounds__(256)
void gemm_wmma(float* __restrict__ out,              // may be nullptr
               const _Float16* __restrict__ wf16,    // [Cout][CinPad] f16
               const float* __restrict__ bias,
               const float* __restrict__ act,        // [B][actCin][NPTS]
               int actCin,
               const float* __restrict__ bcast,      // [B][splitK] or nullptr
               int splitK,
               int Cin, int CinPad, int Cout,
               float* __restrict__ statSum,          // [Cout] (pre-offset)
               float* __restrict__ statSq,
               float* __restrict__ redMax,           // [B][1024] or nullptr
               float* __restrict__ redMin)
{
  const int n0  = blockIdx.x * 128;
  const int m0  = blockIdx.y * 64;
  const int bz  = blockIdx.z;
  const int tid = (int)threadIdx.x;

  __shared__ _Float16 lA[2][64][40];    // 2 x (64x32 W tile), padded
  __shared__ _Float16 lB[2][32][136];   // 2 x (32x128 act tile), padded

  v8f acc[4] = {};

  const int lane = tid & 31;
  const int wv   = tid >> 5;
  const int wm   = (wv & 3) * 16;    // wave M offset inside block tile
  const int wn   = (wv >> 2) * 64;   // wave N offset inside block tile

  // stage K-tile k0 into LDS buffer s
  auto stage = [&](int k0, int s) {
    { // ---- A: weights, already f16, zero-padded at conversion ----
      int e = tid * 8;                     // 256 thr * 8 f16 = 64x32
      int row = e >> 5, col = e & 31;
      uint4 w = *(const uint4*)(wf16 + (size_t)(m0 + row) * CinPad + k0 + col);
      *(uint4*)&lA[s][row][col] = w;
    }
    // ---- B: activations f32 -> f16 ----
#pragma unroll
    for (int i = 0; i < 4; ++i) {
      int idx = i * 1024 + tid * 4;        // 32x128 f32 tile
      int row = idx >> 7, col = idx & 127;
      int k = k0 + row;
      float4 vv;
      if constexpr (GUARD) {
        vv = make_float4(0.f, 0.f, 0.f, 0.f);
        if (k < Cin) {
          if (k < splitK) {                // broadcast global-feature channel
            float gv = bcast[(size_t)bz * splitK + k];
            vv = make_float4(gv, gv, gv, gv);
          } else {
            const float* p = act + ((size_t)bz * actCin + (k - splitK)) * NPTS
                                 + n0 + col;
            vv = *(const float4*)p;
            if (k + 32 < Cin && k + 32 >= splitK)
              __builtin_prefetch(p + 32 * NPTS, 0, 3);
          }
        }
      } else {
        const float* p = act + ((size_t)bz * actCin + k) * NPTS + n0 + col;
        vv = *(const float4*)p;
        if (k0 + 32 < CinPad)              // uniform: prefetch tile after next
          __builtin_prefetch(p + 32 * NPTS, 0, 3);
      }
      H4 h;
      h.h[0] = (_Float16)vv.x; h.h[1] = (_Float16)vv.y;
      h.h[2] = (_Float16)vv.z; h.h[3] = (_Float16)vv.w;
      *(uint2*)&lB[s][row][col] = h.u;
    }
  };

  const int nk = CinPad >> 5;
  stage(0, 0);
  for (int t = 0; t < nk; ++t) {
    __syncthreads();                       // buf[t&1] ready; buf[(t+1)&1] free
    if (t + 1 < nk) stage((t + 1) * 32, (t + 1) & 1);
    const int s = t & 1;

    // ---- A fragment (16x32): lane<16 holds K {0..7,16..23} of row lane,
    //      lane>=16 holds K {8..15,24..31} of row lane-16 (ISA 7.12.2) ----
    FragAB fa;
    {
      int am = lane & 15;
      int kb = (lane < 16) ? 0 : 8;
      const _Float16* pa = &lA[s][wm + am][0];
      fa.u[0] = *(const uint4*)(pa + kb);
      fa.u[1] = *(const uint4*)(pa + kb + 16);
    }
    // ---- B fragments (32x16 each): lane%16 = n col, elements = K {0..7,16..23} / {8..15,24..31} ----
#pragma unroll
    for (int j = 0; j < 4; ++j) {
      FragAB fb;
      const int bkb = (lane < 16) ? 0 : 8;
      const int bn = wn + j * 16 + (lane & 15);
#pragma unroll
      for (int e = 0; e < 8; ++e) {
        fb.v[e] = lB[s][bkb + e][bn];
        fb.v[8 + e] = lB[s][bkb + 16 + e][bn];
      }
      acc[j] = __builtin_amdgcn_wmma_f32_16x16x32_f16(
          false, fa.v, false, fb.v, (short)0, acc[j], false, false);
    }
  }

  // ---- epilogue: bias add, store, fused BN-stat / max-pool reduction ----
  const int half = lane >> 4;          // C/D: vgpr r, lanes16-31 -> M=r+8
  const int lcol = lane & 15;
#pragma unroll
  for (int r = 0; r < 8; ++r) {
    int cg = m0 + wm + r + half * 8;   // global output channel
    float bv = bias ? bias[cg] : 0.f;
    float s = 0.f, ss = 0.f, mx = -3.0e38f, mn = 3.0e38f;
#pragma unroll
    for (int j = 0; j < 4; ++j) {
      float v = acc[j][r] + bv;
      if (out)
        out[((size_t)bz * Cout + cg) * NPTS + n0 + wn + j * 16 + lcol] = v;
      s += v; ss += v * v;
      mx = fmaxf(mx, v); mn = fminf(mn, v);
    }
    // butterfly over the 16 lanes holding this channel's columns
#pragma unroll
    for (int d = 8; d >= 1; d >>= 1) {
      s  += __shfl_xor(s, d);
      ss += __shfl_xor(ss, d);
      mx = fmaxf(mx, __shfl_xor(mx, d));
      mn = fminf(mn, __shfl_xor(mn, d));
    }
    if (lcol == 0) {
      atomicAdd(&statSum[cg], s);
      atomicAdd(&statSq[cg], ss);
      if (redMax) {
        atomicMax(&redMax[(size_t)bz * 1024 + cg], mx);
        atomicMin(&redMin[(size_t)bz * 1024 + cg], mn);
      }
    }
  }
}

// ============================================================================
// Support kernels
// ============================================================================
__global__ void init_stats_kernel(float* __restrict__ ws) {
  int t = blockIdx.x * blockDim.x + threadIdx.x;
  if (t < (int)OFF_MAX)           ws[t] = 0.f;       // sums/sq/scale/shift
  else if (t < (int)OFF_MIN)      ws[t] = -3.0e38f;  // max accum
  else if (t < (int)OFF_HMAX)     ws[t] =  3.0e38f;  // min accum
}

__global__ void convert_w_kernel(_Float16* __restrict__ dst,
                                 const float* __restrict__ src,
                                 int Cout, int Cin, int CinPad) {
  int t = blockIdx.x * blockDim.x + threadIdx.x;
  if (t >= Cout * CinPad) return;
  int o = t / CinPad, i = t % CinPad;
  dst[t] = (_Float16)((i < Cin) ? src[(size_t)o * Cin + i] : 0.f);
}

__global__ void finalize_kernel(const float* __restrict__ sum,
                                const float* __restrict__ sq,
                                float* __restrict__ scale,
                                float* __restrict__ shift,
                                const float* __restrict__ g,
                                const float* __restrict__ bt,
                                int C, float inv) {
  int c = blockIdx.x * blockDim.x + threadIdx.x;
  if (c >= C) return;
  float m = sum[c] * inv;
  float v = sq[c] * inv - m * m;
  float sc = g[c] * rsqrtf(v + 1e-5f);
  scale[c] = sc;
  shift[c] = bt[c] - m * sc;
}

__global__ void bn_relu_kernel(float* __restrict__ buf,
                               const float* __restrict__ scale,
                               const float* __restrict__ shift,
                               int C, long n4) {
  long i = blockIdx.x * (long)blockDim.x + threadIdx.x;   // float4 units
  if (i >= n4) return;
  int c = (int)((i >> 10) % C);                           // 4096/4 per row
  float sc = scale[c], sh = shift[c];
  float4 v = ((float4*)buf)[i];
  v.x = fmaxf(v.x * sc + sh, 0.f);
  v.y = fmaxf(v.y * sc + sh, 0.f);
  v.z = fmaxf(v.z * sc + sh, 0.f);
  v.w = fmaxf(v.w * sc + sh, 0.f);
  ((float4*)buf)[i] = v;
}

// max_n (relu?)(bn(x)) from per-(b,c) max/min of raw conv (BN is monotone
// affine per channel: pick max if scale>=0 else min).
__global__ void pool_kernel(float* __restrict__ dst,
                            const float* __restrict__ mx,
                            const float* __restrict__ mn,
                            const float* __restrict__ scale,
                            const float* __restrict__ shift,
                            int doRelu) {
  int t = blockIdx.x * blockDim.x + threadIdx.x;
  if (t >= 16 * 1024) return;
  int c = t & 1023;
  float sc = scale[c];
  float sel = (sc >= 0.f) ? mx[t] : mn[t];
  float v = sc * sel + shift[c];
  if (doRelu) v = fmaxf(v, 0.f);
  dst[t] = v;
}

__global__ void fc_kernel(float* __restrict__ out,
                          const float* __restrict__ in,
                          const float* __restrict__ w,
                          const float* __restrict__ bias,
                          int Cin, int Cout, int addI) {
  int t = blockIdx.x * blockDim.x + threadIdx.x;
  if (t >= 16 * Cout) return;
  int b = t / Cout, o = t % Cout;
  const float* ip = in + (size_t)b * Cin;
  const float* wp = w + (size_t)o * Cin;
  float s = bias[o];
  for (int i = 0; i < Cin; ++i) s += ip[i] * wp[i];
  if (addI && (o == 0 || o == 4 || o == 8)) s += 1.f;   // + eye(3)
  out[t] = s;
}

__global__ void bn_fc_kernel(float* __restrict__ buf,
                             const float* __restrict__ g,
                             const float* __restrict__ bt, int C) {
  int c = blockIdx.x * blockDim.x + threadIdx.x;
  if (c >= C) return;
  float s = 0.f, ss = 0.f;
  for (int b = 0; b < 16; ++b) { float x = buf[b * C + c]; s += x; ss += x * x; }
  float m = s * (1.f / 16.f);
  float v = ss * (1.f / 16.f) - m * m;
  float sc = g[c] * rsqrtf(v + 1e-5f);
  float sh = bt[c] - m * sc;
  for (int b = 0; b < 16; ++b)
    buf[b * C + c] = fmaxf(buf[b * C + c] * sc + sh, 0.f);
}

// x_new[b,e,n] = sum_d x[b,d,n] * T[b,d,e]
__global__ void transform_kernel(float* __restrict__ xt,
                                 const float* __restrict__ x,
                                 const float* __restrict__ t9) {
  int t = blockIdx.x * blockDim.x + threadIdx.x;
  if (t >= 16 * NPTS) return;
  int b = t >> 12, n = t & 4095;
  const float* xp = x + (size_t)b * 3 * NPTS + n;
  float x0 = xp[0], x1 = xp[NPTS], x2 = xp[2 * NPTS];
  const float* T = t9 + b * 9;
  float* op = xt + (size_t)b * 3 * NPTS + n;
  op[0]        = x0 * T[0] + x1 * T[3] + x2 * T[6];
  op[NPTS]     = x0 * T[1] + x1 * T[4] + x2 * T[7];
  op[2 * NPTS] = x0 * T[2] + x1 * T[5] + x2 * T[8];
}

// head c4 (xo) + c5 (log-softmax over 2 classes) fused; writes xo to ws and
// to both d_out slots, logp to d_out.
__global__ void head_out_kernel(const float* __restrict__ h,   // (16,128,4096)
                                const float* __restrict__ w4,
                                const float* __restrict__ b4,
                                const float* __restrict__ w5,
                                const float* __restrict__ b5,
                                float* __restrict__ xo_ws,
                                float* __restrict__ dout) {
  int t = blockIdx.x * blockDim.x + threadIdx.x;
  if (t >= 16 * NPTS) return;
  int b = t >> 12, n = t & 4095;
  const float* hp = h + (size_t)b * 128 * NPTS + n;
  float u0 = b4[0], u1 = b4[1], v0 = b5[0], v1 = b5[1];
#pragma unroll 4
  for (int i = 0; i < 128; ++i) {
    float x = hp[(size_t)i * NPTS];
    u0 += w4[i] * x;       u1 += w4[128 + i] * x;
    v0 += w5[i] * x;       v1 += w5[128 + i] * x;
  }
  size_t o2 = (size_t)t * 2;
  xo_ws[o2] = u0;             xo_ws[o2 + 1] = u1;
  dout[OUT_XO1 + o2] = u0;    dout[OUT_XO1 + o2 + 1] = u1;
  dout[OUT_XO2 + o2] = u0;    dout[OUT_XO2 + o2 + 1] = u1;
  float m = fmaxf(v0, v1);
  float l = m + logf(expf(v0 - m) + expf(v1 - m));
  dout[OUT_LOGP + o2]     = v0 - l;
  dout[OUT_LOGP + o2 + 1] = v1 - l;
}

__device__ __forceinline__ float ang2(float ax, float ay, float bx, float by,
                                      float cx, float cy) {
  float e1x = bx - ax, e1y = by - ay, e2x = cx - ax, e2y = cy - ay;
  float num = e1x * e2x + e1y * e2y;
  float den = sqrtf(e1x * e1x + e1y * e1y) * sqrtf(e2x * e2x + e2y * e2y)
            + 1e-8f;
  return num / den;
}

__global__ void face_kernel(const int* __restrict__ faces,
                            const float* __restrict__ xo,
                            float* __restrict__ dout) {
  int t = blockIdx.x * blockDim.x + threadIdx.x;
  if (t >= 16 * NFACE) return;
  int b = t >> 13;
  const int* fp = faces + (size_t)t * 3;
  const float* xb = xo + (size_t)b * NPTS * 2;
  int i0 = fp[0], i1 = fp[1], i2 = fp[2];
  float x1 = xb[i0 * 2], y1 = xb[i0 * 2 + 1];
  float x2 = xb[i1 * 2], y2 = xb[i1 * 2 + 1];
  float x3 = xb[i2 * 2], y3 = xb[i2 * 2 + 1];
  float area = 0.5f * (x1 * (y2 - y3) + x2 * (y3 - y1) + x3 * (y1 - y2));
  float c0 = ang2(x1, y1, x2, y2, x3, y3);
  float c1 = ang2(x2, y2, x3, y3, x1, y1);
  float c2 = ang2(x3, y3, x1, y1, x2, y2);
  dout[OUT_AREA1 + t] = area;
  dout[OUT_AREA2 + t] = area;
  size_t c = OUT_COS1 + (size_t)t * 3;
  dout[c] = c0; dout[c + 1] = c1; dout[c + 2] = c2;
  c = OUT_COS2 + (size_t)t * 3;
  dout[c] = c0; dout[c + 1] = c1; dout[c + 2] = c2;
}

// ============================================================================
// Host launcher
// ============================================================================
extern "C" void kernel_launch(void* const* d_in, const int* in_sizes, int n_in,
                              void* d_out, int out_size, void* d_ws,
                              size_t ws_size, hipStream_t stream) {
  (void)in_sizes; (void)n_in; (void)out_size; (void)ws_size;

  float* ws   = (float*)d_ws;
  float* dout = (float*)d_out;
  const float* x_in  = (const float*)d_in[IN_X];
  const int*   faces = (const int*)d_in[IN_FACES];
  auto P = [&](int i) { return (const float*)d_in[i]; };

  _Float16* wf = (_Float16*)(ws + OFF_WF16);
  float* buf64  = ws + OFF_B64;
  float* buf128 = ws + OFF_B128;
  float* bufE   = ws + OFF_BE;     // head b1, then reused as head b3
  float* bufF   = ws + OFF_BF;
  float* xt     = ws + OFF_XT;
  float* hmax   = ws + OFF_HMAX;
  float* fc1b   = ws + OFF_FC1;
  float* fc2b   = ws + OFF_FC2;
  float* t9     = ws + OFF_T9;
  float* gfeat  = ws + OFF_G;
  float* xo_ws  = ws + OFF_XOWS;

  // reset BN-stat and max/min accumulators (fresh every call: deterministic)
  init_stats_kernel<<<(unsigned)((OFF_HMAX + 255) / 256), 256, 0, stream>>>(ws);

  // convert + zero-pad all WMMA weights to f16 once per call
  auto convw = [&](size_t off, int pw, int Cin, int Cout) {
    int CinPad = (Cin + 31) & ~31;
    int total = Cout * CinPad;
    convert_w_kernel<<<(total + 255) / 256, 256, 0, stream>>>(
        wf + off, P(pw), Cout, Cin, CinPad);
  };
  convw(WO_STN_C1,  P_STN_C1_W,  3,   64);
  convw(WO_STN_C2,  P_STN_C2_W,  64,  128);
  convw(WO_STN_C3,  P_STN_C3_W,  128, 1024);
  convw(WO_FEAT_C1, P_FEAT_C1_W, 3,   64);
  convw(WO_FEAT_C2, P_FEAT_C2_W, 64,  128);
  convw(WO_FEAT_C3, P_FEAT_C3_W, 128, 1024);
  convw(WO_HEAD_C1, P_HEAD_C1_W, 1088, 512);
  convw(WO_HEAD_C2, P_HEAD_C2_W, 512, 256);
  convw(WO_HEAD_C3, P_HEAD_C3_W, 256, 128);

  auto gemm = [&](float* out, size_t woff, int pbias, const float* act,
                  int actCin, const float* bcast, int splitK, int Cin,
                  int Cout, int slot, int redslot) {
    int CinPad = (Cin + 31) & ~31;
    dim3 g(NPTS / 128, Cout / 64, BATCH);
    float* sum = ws + OFF_SUM + (size_t)slot * 1024;
    float* sq  = ws + OFF_SQ  + (size_t)slot * 1024;
    float* rmx = redslot >= 0 ? ws + OFF_MAX + (size_t)redslot * 16384 : nullptr;
    float* rmn = redslot >= 0 ? ws + OFF_MIN + (size_t)redslot * 16384 : nullptr;
    bool fast = (Cin == CinPad) && (splitK == 0);
    if (fast)
      gemm_wmma<false><<<g, 256, 0, stream>>>(
          out, wf + woff, P(pbias), act, actCin, bcast, splitK, Cin, CinPad,
          Cout, sum, sq, rmx, rmn);
    else
      gemm_wmma<true><<<g, 256, 0, stream>>>(
          out, wf + woff, P(pbias), act, actCin, bcast, splitK, Cin, CinPad,
          Cout, sum, sq, rmx, rmn);
  };
  auto fin = [&](int slot, int pg, int pb, int C) {
    finalize_kernel<<<(C + 255) / 256, 256, 0, stream>>>(
        ws + OFF_SUM + (size_t)slot * 1024, ws + OFF_SQ + (size_t)slot * 1024,
        ws + OFF_SCALE + (size_t)slot * 1024,
        ws + OFF_SHIFT + (size_t)slot * 1024, P(pg), P(pb), C,
        1.f / (float)(BATCH * NPTS));
  };
  auto bnr = [&](float* buf, int slot, int C) {
    long n4 = (long)BATCH * C * NPTS / 4;
    bn_relu_kernel<<<(unsigned)((n4 + 255) / 256), 256, 0, stream>>>(
        buf, ws + OFF_SCALE + (size_t)slot * 1024,
        ws + OFF_SHIFT + (size_t)slot * 1024, C, n4);
  };

  // ---------------- STN ----------------
  gemm(buf64, WO_STN_C1, P_STN_C1_B, x_in, 3, nullptr, 0, 3, 64, 0, -1);
  fin(0, P_STN_BN1_G, P_STN_BN1_B, 64);  bnr(buf64, 0, 64);
  gemm(buf128, WO_STN_C2, P_STN_C2_B, buf64, 64, nullptr, 0, 64, 128, 1, -1);
  fin(1, P_STN_BN2_G, P_STN_BN2_B, 128); bnr(buf128, 1, 128);
  gemm(nullptr, WO_STN_C3, P_STN_C3_B, buf128, 128, nullptr, 0, 128, 1024,
       2, 0);                                   // fused stats+max, no store
  fin(2, P_STN_BN3_G, P_STN_BN3_B, 1024);
  pool_kernel<<<64, 256, 0, stream>>>(hmax, ws + OFF_MAX, ws + OFF_MIN,
                                      ws + OFF_SCALE + 2 * 1024,
                                      ws + OFF_SHIFT + 2 * 1024, 1);
  fc_kernel<<<(16 * 512 + 255) / 256, 256, 0, stream>>>(
      fc1b, hmax, P(P_STN_FC1_W), P(P_STN_FC1_B), 1024, 512, 0);
  bn_fc_kernel<<<2, 256, 0, stream>>>(fc1b, P(P_STN_BN4_G), P(P_STN_BN4_B), 512);
  fc_kernel<<<(16 * 256 + 255) / 256, 256, 0, stream>>>(
      fc2b, fc1b, P(P_STN_FC2_W), P(P_STN_FC2_B), 512, 256, 0);
  bn_fc_kernel<<<1, 256, 0, stream>>>(fc2b, P(P_STN_BN5_G), P(P_STN_BN5_B), 256);
  fc_kernel<<<1, 256, 0, stream>>>(t9, fc2b, P(P_STN_FC3_W), P(P_STN_FC3_B),
                                   256, 9, 1);
  transform_kernel<<<(16 * NPTS + 255) / 256, 256, 0, stream>>>(xt, x_in, t9);

  // ---------------- feature net ----------------
  gemm(buf64, WO_FEAT_C1, P_FEAT_C1_B, xt, 3, nullptr, 0, 3, 64, 3, -1);
  fin(3, P_FEAT_BN1_G, P_FEAT_BN1_B, 64);  bnr(buf64, 3, 64);   // = pointfeat
  gemm(buf128, WO_FEAT_C2, P_FEAT_C2_B, buf64, 64, nullptr, 0, 64, 128, 4, -1);
  fin(4, P_FEAT_BN2_G, P_FEAT_BN2_B, 128); bnr(buf128, 4, 128);
  gemm(nullptr, WO_FEAT_C3, P_FEAT_C3_B, buf128, 128, nullptr, 0, 128, 1024,
       5, 1);
  fin(5, P_FEAT_BN3_G, P_FEAT_BN3_B, 1024);
  pool_kernel<<<64, 256, 0, stream>>>(gfeat, ws + OFF_MAX + 16384,
                                      ws + OFF_MIN + 16384,
                                      ws + OFF_SCALE + 5 * 1024,
                                      ws + OFF_SHIFT + 5 * 1024, 0); // no relu

  // ---------------- head (concat [g(1024) ; pointfeat(64)] on the fly) ----
  gemm(bufE, WO_HEAD_C1, P_HEAD_C1_B, buf64, 64, gfeat, 1024, 1088, 512,
       6, -1);
  fin(6, P_HEAD_BN1_G, P_HEAD_BN1_B, 512); bnr(bufE, 6, 512);
  gemm(bufF, WO_HEAD_C2, P_HEAD_C2_B, bufE, 512, nullptr, 0, 512, 256, 7, -1);
  fin(7, P_HEAD_BN2_G, P_HEAD_BN2_B, 256); bnr(bufF, 7, 256);
  gemm(bufE, WO_HEAD_C3, P_HEAD_C3_B, bufF, 256, nullptr, 0, 256, 128, 8, -1);
  fin(8, P_HEAD_BN3_G, P_HEAD_BN3_B, 128); bnr(bufE, 8, 128);

  head_out_kernel<<<(16 * NPTS + 255) / 256, 256, 0, stream>>>(
      bufE, P(P_HEAD_C4_W), P(P_HEAD_C4_B), P(P_HEAD_C5_W), P(P_HEAD_C5_B),
      xo_ws, dout);
  face_kernel<<<(16 * NFACE + 255) / 256, 256, 0, stream>>>(faces, xo_ws, dout);
}
